// Attention_70574902608507
// MI455X (gfx1250) — compile-verified, matrix operand order checked by reading, not run on hardware
//
#include <hip/hip_runtime.h>
#include <hip/hip_bf16.h>
#include <math.h>

typedef __attribute__((ext_vector_type(16))) __bf16 v16bf;
typedef __attribute__((ext_vector_type(8)))  float  v8f;

#define BM 128
#define BN 128
#define BK 32

#if __has_builtin(__builtin_amdgcn_global_load_async_to_lds_b128)
#define HAVE_ASYNC_LDS 1
#else
#define HAVE_ASYNC_LDS 0
#endif

#if HAVE_ASYNC_LDS
typedef int v4i_async __attribute__((vector_size(16)));
typedef __attribute__((address_space(1))) v4i_async* gasync_p;
typedef __attribute__((address_space(3))) v4i_async* lasync_p;
#endif

// 16-byte global -> LDS copy. Async (ASYNCcnt-tracked, no VGPR bounce) when
// the CDNA5 builtin is available; plain copy otherwise.
__device__ __forceinline__ void g2l16(unsigned short* l, const unsigned short* g) {
#if HAVE_ASYNC_LDS
  __builtin_amdgcn_global_load_async_to_lds_b128((gasync_p)g, (lasync_p)l, 0, 0);
#else
  *(uint4*)l = *(const uint4*)g;
#endif
}

__device__ __forceinline__ void async_wait0() {
#if HAVE_ASYNC_LDS
#if __has_builtin(__builtin_amdgcn_s_wait_asynccnt)
  __builtin_amdgcn_s_wait_asynccnt(0);
#else
  asm volatile("s_wait_asynccnt 0x0" ::: "memory");
#endif
#endif
}

__device__ __forceinline__ unsigned short f2bf(float f) {
  union { float f; unsigned int u; } c;
  c.f = f;
  unsigned int u = c.u + 0x7fffu + ((c.u >> 16) & 1u);
  return (unsigned short)(u >> 16);
}

struct Frag {
  union { v16bf v; unsigned int u[8]; };
};

// Build a 16x32 bf16 WMMA operand fragment from an LDS tile stored row-major
// ([16 rows] x [ldk ushorts]).  Matches CDNA5 16-bit A-matrix layout:
// lane = row%16 | hi-half, VGPR j holds K = (j<4?0:16) + 8*hi + 2*(j%4) +{0,1}.
__device__ __forceinline__ void load_frag(Frag& f, const unsigned short* sub, int ldk) {
  const int lane = threadIdx.x & 31;
  const int hi = lane >> 4;
  const unsigned int* p = (const unsigned int*)(sub + (lane & 15) * ldk);
#pragma unroll
  for (int j = 0; j < 8; ++j) {
    const int k = ((j & 4) << 2) + hi * 8 + ((j & 3) << 1);
    f.u[j] = p[k >> 1];
  }
}

// Transpose-scatter a 32x128 bf16 B tile (held in 2 x uint4 regs) into sB[n][k].
__device__ __forceinline__ void scatterB(unsigned short* sB, const uint4* vb, int tid) {
#pragma unroll
  for (int i = 0; i < 2; ++i) {
    const int l = tid + 256 * i;
    const int k = l >> 4;
    const int ncg = (l & 15) << 3;
    const unsigned int w[4] = {vb[i].x, vb[i].y, vb[i].z, vb[i].w};
#pragma unroll
    for (int j = 0; j < 8; ++j)
      sB[(ncg + j) * BK + k] = (unsigned short)(w[j >> 1] >> ((j & 1) * 16));
  }
}

// ---------------------------------------------------------------------------
// LayerNorm: one 256-thread block per row (H = 2048 -> 8 elems/thread), bf16 out
// ---------------------------------------------------------------------------
__global__ __launch_bounds__(256) void ln_kernel(const float* __restrict__ x,
                                                 unsigned short* __restrict__ xn,
                                                 int H) {
  __shared__ float red[256];
  const int row = blockIdx.x;
  const int tid = threadIdx.x;
  const float* xr = x + (long long)row * H;
  float v[8];
  float s = 0.f;
#pragma unroll
  for (int i = 0; i < 8; ++i) { v[i] = xr[tid + 256 * i]; s += v[i]; }
  red[tid] = s;
  __syncthreads();
  for (int o = 128; o; o >>= 1) {
    if (tid < o) red[tid] += red[tid + o];
    __syncthreads();
  }
  const float mu = red[0] * (1.0f / 2048.0f);
  __syncthreads();
  float vs = 0.f;
#pragma unroll
  for (int i = 0; i < 8; ++i) { float d = v[i] - mu; vs += d * d; }
  red[tid] = vs;
  __syncthreads();
  for (int o = 128; o; o >>= 1) {
    if (tid < o) red[tid] += red[tid + o];
    __syncthreads();
  }
  const float rs = rsqrtf(red[0] * (1.0f / 2048.0f) + 1e-6f);
  unsigned short* xo = xn + (long long)row * H;
#pragma unroll
  for (int i = 0; i < 8; ++i) xo[tid + 256 * i] = f2bf((v[i] - mu) * rs);
}

// ---------------------------------------------------------------------------
// fp32 -> bf16 weight conversion (grid-stride)
// ---------------------------------------------------------------------------
__global__ void cvt_kernel(const float* __restrict__ in,
                           unsigned short* __restrict__ outp, long long n) {
  long long i = (long long)blockIdx.x * blockDim.x + threadIdx.x;
  const long long stride = (long long)gridDim.x * blockDim.x;
  for (; i < n; i += stride) outp[i] = f2bf(in[i]);
}

// ---------------------------------------------------------------------------
// Generic bf16 WMMA GEMM: C = A @ B  (A [M,K] row-major, B [K,N] row-major)
// 256 threads = 8 waves in 2x4; block tile 128x128, K step 32.
// Double-buffered LDS; A tile via async-to-LDS, B tile pipelined via regs.
// OUT_MODE 0: bf16 C.  OUT_MODE 1: fp32 C = acc + Res (residual add).
// CAUSAL: limit K loop to m0+BM (lower-triangular A support for P@V).
// ---------------------------------------------------------------------------
template <int OUT_MODE, bool CAUSAL>
__global__ __launch_bounds__(256) void gemm_bf16_kernel(
    const unsigned short* __restrict__ A, long long strideA, int lda,
    const unsigned short* __restrict__ Bmat, long long strideB, int ldb,
    void* __restrict__ C, long long strideC, int ldc,
    const float* __restrict__ Res, int K) {
  __shared__ unsigned short sA[2][BM * BK];
  __shared__ unsigned short sB[2][BN * BK];  // stored transposed: sB[n][k]

  const int tid = threadIdx.x;
  const int wave = tid >> 5;
  const int wm = wave & 1;   // 64-row block
  const int wn = wave >> 1;  // 32-col block
  const int m0 = blockIdx.y * BM;
  const int n0 = blockIdx.x * BN;
  const int batch = blockIdx.z;

  const unsigned short* Ab = A + (long long)batch * strideA;
  const unsigned short* Bb = Bmat + (long long)batch * strideB;

  const v8f vzero = {0.f, 0.f, 0.f, 0.f, 0.f, 0.f, 0.f, 0.f};
  v8f acc[4][2];
#pragma unroll
  for (int mi = 0; mi < 4; ++mi)
#pragma unroll
    for (int ni = 0; ni < 2; ++ni) acc[mi][ni] = vzero;

  const int kEnd = CAUSAL ? ((m0 + BM) < K ? (m0 + BM) : K) : K;
  const int nt = kEnd / BK;

  // ---- prologue: stage tile 0 into buffer 0 ----
#pragma unroll
  for (int i = 0; i < 2; ++i) {
    const int l = tid + 256 * i;
    const int r = l >> 2;
    const int cg = (l & 3) << 3;
    g2l16(&sA[0][r * BK + cg], Ab + (long long)(m0 + r) * lda + cg);
  }
  {
    uint4 vb[2];
#pragma unroll
    for (int i = 0; i < 2; ++i) {
      const int l = tid + 256 * i;
      vb[i] = *(const uint4*)(Bb + (long long)(l >> 4) * ldb + n0 + ((l & 15) << 3));
    }
    scatterB(sB[0], vb, tid);
  }

  int cur = 0;
  for (int t = 0; t < nt; ++t) {
    async_wait0();        // current buffer's async A data has landed in LDS
    __syncthreads();      // all waves done with the other buffer; staging visible

    const bool havenext = (t + 1 < nt);
    uint4 nb[2];
    if (havenext) {
      const int ktn = (t + 1) * BK;
#pragma unroll
      for (int i = 0; i < 2; ++i) {
        const int l = tid + 256 * i;
        const int r = l >> 2;
        const int cg = (l & 3) << 3;
        g2l16(&sA[cur ^ 1][r * BK + cg],
              Ab + (long long)(m0 + r) * lda + ktn + cg);
      }
#pragma unroll
      for (int i = 0; i < 2; ++i) {
        const int l = tid + 256 * i;
        nb[i] = *(const uint4*)(Bb + (long long)(ktn + (l >> 4)) * ldb + n0 +
                                ((l & 15) << 3));
      }
    }

    Frag af[4], bfr[2];
#pragma unroll
    for (int mi = 0; mi < 4; ++mi)
      load_frag(af[mi], &sA[cur][(wm * 64 + mi * 16) * BK], BK);
#pragma unroll
    for (int ni = 0; ni < 2; ++ni)
      load_frag(bfr[ni], &sB[cur][(wn * 32 + ni * 16) * BK], BK);
#pragma unroll
    for (int mi = 0; mi < 4; ++mi)
#pragma unroll
      for (int ni = 0; ni < 2; ++ni)
        acc[mi][ni] = __builtin_amdgcn_wmma_f32_16x16x32_bf16(
            false, af[mi].v, false, bfr[ni].v, (short)0, acc[mi][ni], false,
            false);

    if (havenext) scatterB(sB[cur ^ 1], nb, tid);  // overlaps loadcnt wait w/ WMMA
    cur ^= 1;
  }

  // Epilogue. C/D layout: lane L, vgpr r -> m = r + 8*(L>>4), n = L & 15.
  const int lane = tid & 31;
  const int hi = lane >> 4;
  const int nn = lane & 15;
  if (OUT_MODE == 0) {
    unsigned short* Cb = (unsigned short*)C + (long long)batch * strideC;
#pragma unroll
    for (int mi = 0; mi < 4; ++mi)
#pragma unroll
      for (int ni = 0; ni < 2; ++ni)
#pragma unroll
        for (int r = 0; r < 8; ++r) {
          const int gm = m0 + wm * 64 + mi * 16 + r + 8 * hi;
          const int gn = n0 + wn * 32 + ni * 16 + nn;
          Cb[(long long)gm * ldc + gn] = f2bf(acc[mi][ni][r]);
        }
  } else {
    float* Cb = (float*)C + (long long)batch * strideC;
#pragma unroll
    for (int mi = 0; mi < 4; ++mi)
#pragma unroll
      for (int ni = 0; ni < 2; ++ni)
#pragma unroll
        for (int r = 0; r < 8; ++r) {
          const int gm = m0 + wm * 64 + mi * 16 + r + 8 * hi;
          const int gn = n0 + wn * 32 + ni * 16 + nn;
          Cb[(long long)gm * ldc + gn] =
              acc[mi][ni][r] + Res[(long long)gm * ldc + gn];
        }
  }
}

// ---------------------------------------------------------------------------
// Causal attention scores, two passes sharing one kernel body:
//  WRITE_PROBS=false: compute per-row softmax stats (max m, sumexp l) online.
//  WRITE_PROBS=true : recompute scores, normalize with (m,l), emit bf16 probs.
// Q/K are bf16 slices of qkv_out ([B*S, 3D], row stride 3D).
// Block = 128 query rows; streams 128-key tiles up to the causal limit.
// Both Q and K tiles staged with async-to-LDS, double buffered over the D loop.
// ---------------------------------------------------------------------------
template <bool WRITE_PROBS>
__global__ __launch_bounds__(256) void attn_score_kernel(
    const unsigned short* __restrict__ qkv, float* __restrict__ statsM,
    float* __restrict__ statsL, unsigned short* __restrict__ probs, int S,
    int D, float scale) {
  __shared__ unsigned short sQ[2][BM * BK];
  __shared__ unsigned short sK[2][BN * BK];
  __shared__ float pRed[4][BM];
  __shared__ float cM[BM];
  __shared__ float rowM[BM];
  __shared__ float rowL[BM];

  const int tid = threadIdx.x;
  const int wave = tid >> 5;
  const int wm = wave & 1;
  const int wn = wave >> 1;
  const int lane = tid & 31;
  const int hi = lane >> 4;
  const int nn = lane & 15;
  const int b = blockIdx.y;
  const int m0 = blockIdx.x * BM;
  const int ld3 = 3 * D;
  const unsigned short* Q = qkv + (long long)b * S * ld3;
  const unsigned short* Km = Q + D;

  if (tid < BM) {
    if (WRITE_PROBS) {
      rowM[tid] = statsM[(long long)b * S + m0 + tid];
      rowL[tid] = 1.0f / statsL[(long long)b * S + m0 + tid];  // inverse l
    } else {
      rowM[tid] = -__builtin_inff();
      rowL[tid] = 0.f;
    }
  }

  const v8f vzero = {0.f, 0.f, 0.f, 0.f, 0.f, 0.f, 0.f, 0.f};
  const int kTileEnd = m0 + BM;  // causal: keys < m0 + 128
  const int ndt = D / BK;

  for (int kt0 = 0; kt0 < kTileEnd; kt0 += BN) {
    __syncthreads();  // previous tile's reads + pRed use complete

    // prologue: stage d-chunk 0 into buffer 0
#pragma unroll
    for (int i = 0; i < 2; ++i) {
      const int l = tid + 256 * i;
      const int r = l >> 2;
      const int cg = (l & 3) << 3;
      g2l16(&sQ[0][r * BK + cg], Q + (long long)(m0 + r) * ld3 + cg);
      g2l16(&sK[0][r * BK + cg], Km + (long long)(kt0 + r) * ld3 + cg);
    }

    v8f acc[4][2];
#pragma unroll
    for (int mi = 0; mi < 4; ++mi)
#pragma unroll
      for (int ni = 0; ni < 2; ++ni) acc[mi][ni] = vzero;

    int cur = 0;
    for (int dtt = 0; dtt < ndt; ++dtt) {
      async_wait0();
      __syncthreads();
      if (dtt + 1 < ndt) {
        const int dtn = (dtt + 1) * BK;
#pragma unroll
        for (int i = 0; i < 2; ++i) {
          const int l = tid + 256 * i;
          const int r = l >> 2;
          const int cg = (l & 3) << 3;
          g2l16(&sQ[cur ^ 1][r * BK + cg],
                Q + (long long)(m0 + r) * ld3 + dtn + cg);
          g2l16(&sK[cur ^ 1][r * BK + cg],
                Km + (long long)(kt0 + r) * ld3 + dtn + cg);
        }
      }
      Frag aq[4], bk[2];
#pragma unroll
      for (int mi = 0; mi < 4; ++mi)
        load_frag(aq[mi], &sQ[cur][(wm * 64 + mi * 16) * BK], BK);
#pragma unroll
      for (int ni = 0; ni < 2; ++ni)
        load_frag(bk[ni], &sK[cur][(wn * 32 + ni * 16) * BK], BK);
#pragma unroll
      for (int mi = 0; mi < 4; ++mi)
#pragma unroll
        for (int ni = 0; ni < 2; ++ni)
          acc[mi][ni] = __builtin_amdgcn_wmma_f32_16x16x32_bf16(
              false, aq[mi].v, false, bk[ni].v, (short)0, acc[mi][ni], false,
              false);
      cur ^= 1;
    }

    if (!WRITE_PROBS) {
      // --- online stats update: per-row max over this 128-key tile ---
#pragma unroll
      for (int mi = 0; mi < 4; ++mi)
#pragma unroll
        for (int r = 0; r < 8; ++r) {
          const int mloc = wm * 64 + mi * 16 + r + 8 * hi;
          const int qI = m0 + mloc;
          float v = -__builtin_inff();
#pragma unroll
          for (int ni = 0; ni < 2; ++ni) {
            const int kI = kt0 + wn * 32 + ni * 16 + nn;
            const float s =
                (kI <= qI) ? acc[mi][ni][r] * scale : -__builtin_inff();
            v = fmaxf(v, s);
          }
#pragma unroll
          for (int off = 8; off; off >>= 1)
            v = fmaxf(v, __shfl_xor(v, off, 16));
          if (nn == 0) pRed[wn][mloc] = v;
        }
      __syncthreads();
      if (tid < BM) {
        const float t = fmaxf(fmaxf(pRed[0][tid], pRed[1][tid]),
                              fmaxf(pRed[2][tid], pRed[3][tid]));
        const float old = rowM[tid];
        const float nm = fmaxf(old, t);
        cM[tid] = nm;
        rowL[tid] = rowL[tid] * __expf(old - nm);
        rowM[tid] = nm;
      }
      __syncthreads();
      // --- per-row sumexp contribution ---
#pragma unroll
      for (int mi = 0; mi < 4; ++mi)
#pragma unroll
        for (int r = 0; r < 8; ++r) {
          const int mloc = wm * 64 + mi * 16 + r + 8 * hi;
          const int qI = m0 + mloc;
          const float mrow = cM[mloc];
          float v = 0.f;
#pragma unroll
          for (int ni = 0; ni < 2; ++ni) {
            const int kI = kt0 + wn * 32 + ni * 16 + nn;
            const float s =
                (kI <= qI) ? acc[mi][ni][r] * scale : -__builtin_inff();
            v += __expf(s - mrow);
          }
#pragma unroll
          for (int off = 8; off; off >>= 1) v += __shfl_xor(v, off, 16);
          if (nn == 0) pRed[wn][mloc] = v;
        }
      __syncthreads();
      if (tid < BM)
        rowL[tid] += pRed[0][tid] + pRed[1][tid] + pRed[2][tid] + pRed[3][tid];
      // next key tile's leading __syncthreads orders pRed reuse
    } else {
      // --- emit normalized bf16 probabilities ---
      unsigned short* Pb = probs + (long long)b * S * S;
#pragma unroll
      for (int mi = 0; mi < 4; ++mi)
#pragma unroll
        for (int ni = 0; ni < 2; ++ni)
#pragma unroll
          for (int r = 0; r < 8; ++r) {
            const int mloc = wm * 64 + mi * 16 + r + 8 * hi;
            const int qI = m0 + mloc;
            const int kI = kt0 + wn * 32 + ni * 16 + nn;
            const float s =
                (kI <= qI) ? acc[mi][ni][r] * scale : -__builtin_inff();
            const float p = __expf(s - rowM[mloc]) * rowL[mloc];
            Pb[(long long)qI * S + kI] = f2bf(p);
          }
    }
  }

  if (!WRITE_PROBS && tid < BM) {
    statsM[(long long)b * S + m0 + tid] = rowM[tid];
    statsL[(long long)b * S + m0 + tid] = rowL[tid];
  }
}

// ---------------------------------------------------------------------------
extern "C" void kernel_launch(void* const* d_in, const int* in_sizes, int n_in,
                              void* d_out, int out_size, void* d_ws,
                              size_t ws_size, hipStream_t stream) {
  (void)in_sizes; (void)n_in; (void)out_size; (void)ws_size;
  const int Bb = 2, S = 4096, H = 2048, D = 2048;
  const long long BS = (long long)Bb * S;  // 8192 rows

  const float* x = (const float*)d_in[0];
  const float* qkvw = (const float*)d_in[1];
  const float* ow = (const float*)d_in[2];
  float* out = (float*)d_out;

  char* p = (char*)d_ws;
  auto take = [&](size_t bytes) {
    char* r = p;
    p += (bytes + 255) & ~(size_t)255;
    return r;
  };
  unsigned short* xn = (unsigned short*)take((size_t)BS * H * 2);
  unsigned short* wqkv = (unsigned short*)take((size_t)H * 3 * D * 2);
  unsigned short* wo = (unsigned short*)take((size_t)D * H * 2);
  unsigned short* qkvo = (unsigned short*)take((size_t)BS * 3 * D * 2);
  unsigned short* probs = (unsigned short*)take((size_t)Bb * S * S * 2);
  unsigned short* attn = (unsigned short*)take((size_t)BS * D * 2);
  float* statsM = (float*)take((size_t)BS * 4);
  float* statsL = (float*)take((size_t)BS * 4);

  // 1) LayerNorm -> bf16 x_norm ; weight conversions
  ln_kernel<<<(int)BS, 256, 0, stream>>>(x, xn, H);
  cvt_kernel<<<512, 256, 0, stream>>>(qkvw, wqkv, (long long)H * 3 * D);
  cvt_kernel<<<512, 256, 0, stream>>>(ow, wo, (long long)D * H);

  // 2) qkv_out = x_norm @ Wqkv   (M=8192, N=6144, K=2048) -> bf16
  gemm_bf16_kernel<0, false><<<dim3(3 * D / BN, BS / BM, 1), 256, 0, stream>>>(
      xn, 0, H, wqkv, 0, 3 * D, (void*)qkvo, 0, 3 * D, nullptr, H);

  // 3) softmax stats, then 4) normalized bf16 probs (causal)
  const float scale = 1.0f / sqrtf((float)D);
  attn_score_kernel<false><<<dim3(S / BM, Bb), 256, 0, stream>>>(
      qkvo, statsM, statsL, nullptr, S, D, scale);
  attn_score_kernel<true><<<dim3(S / BM, Bb), 256, 0, stream>>>(
      qkvo, statsM, statsL, probs, S, D, scale);

  // 5) attn = probs @ V  (per batch M=4096, N=2048, K<=m0+128 causal) -> bf16
  gemm_bf16_kernel<0, true><<<dim3(D / BN, S / BM, Bb), 256, 0, stream>>>(
      probs, (long long)S * S, S, qkvo + 2 * D, (long long)S * 3 * D, 3 * D,
      (void*)attn, (long long)S * D, D, nullptr, S);

  // 6) out = attn @ Wo + residual  (M=8192, N=2048, K=2048) -> fp32
  gemm_bf16_kernel<1, false><<<dim3(H / BN, BS / BM, 1), 256, 0, stream>>>(
      attn, 0, D, wo, 0, H, (void*)out, 0, H, x, D);
}
